// SlidingWindowCausalAttention_8744553415139
// MI455X (gfx1250) — compile-verified, matrix operand order checked by reading, not run on hardware
//
#include <hip/hip_runtime.h>
#include <math.h>

#define T_SEQ 2048
#define C_DIM 1024
#define H_NUM 16
#define D_HEAD 64
#define WIN 128

typedef __bf16 bf16_t;
typedef __bf16 v16bf __attribute__((ext_vector_type(16)));
typedef __bf16 v8bf  __attribute__((ext_vector_type(8)));
typedef float  v8f   __attribute__((ext_vector_type(8)));
typedef float  v4f   __attribute__((ext_vector_type(4)));

typedef __attribute__((address_space(3))) bf16_t lds_bf16;

static __device__ __forceinline__ v16bf cat16(v8bf lo, v8bf hi) {
  return __builtin_shufflevector(lo, hi, 0,1,2,3,4,5,6,7,8,9,10,11,12,13,14,15);
}
static __device__ __forceinline__ v8bf zero8() {
  v8bf z;
#pragma unroll
  for (int i = 0; i < 8; ++i) z[i] = (bf16_t)0.0f;
  return z;
}
// One lane's share of a 16x32 (A) / 32x16 (B) bf16 WMMA fragment (ISA 7.12.2):
// lane<16: K {0..7, 16..23}; lane>=16: K {8..15, 24..31}  (p pre-offset by k0).
static __device__ __forceinline__ v16bf load_frag(const bf16_t* p) {
  v8bf lo = *(const v8bf*)p;
  v8bf hi = *(const v8bf*)(p + 16);
  return cat16(lo, hi);
}
static __device__ __forceinline__ v8f wmma_bf16(v16bf a, v16bf b, v8f c) {
  return __builtin_amdgcn_wmma_f32_16x16x32_bf16(false, a, false, b, (short)0, c, false, false);
}

// Async DMA of 16 bytes global -> LDS, tracked with ASYNCcnt (ISA 10.7 / 15.18.3).
static __device__ __forceinline__ void async_copy_b128(const bf16_t* gsrc, bf16_t* ldst) {
  unsigned lds_off = (unsigned)(uintptr_t)(lds_bf16*)ldst;           // AS3 byte offset
  unsigned long long gaddr = (unsigned long long)(uintptr_t)gsrc;
  asm volatile("global_load_async_to_lds_b128 %0, %1, off"
               :: "v"(lds_off), "v"(gaddr) : "memory");
}
static __device__ __forceinline__ void wait_asynccnt_le2() {
  asm volatile("s_wait_asynccnt 0x2" ::: "memory");
}
static __device__ __forceinline__ void wait_asynccnt_0() {
  asm volatile("s_wait_asynccnt 0x0" ::: "memory");
}

// ---------------------------------------------------------------- convert
__global__ void __launch_bounds__(256)
f32_to_bf16_kernel(const float* __restrict__ in, bf16_t* __restrict__ out, int n) {
  int i = (blockIdx.x * 256 + threadIdx.x) * 8;
  if (i < n) {
    v4f a = *(const v4f*)(in + i);
    v4f b = *(const v4f*)(in + i + 4);
    v8bf o;
#pragma unroll
    for (int j = 0; j < 4; ++j) { o[j] = (bf16_t)a[j]; o[j + 4] = (bf16_t)b[j]; }
    *(v8bf*)(out + i) = o;
  }
}

// ---------------------------------------------------------------- GEMM
// Out[M,N] = A[M,K] * B[N,K]^T + bias[N]; A,B bf16 row-major.
// Block = 128 threads (4 waves); block tile 128(M) x 64(N); wave tile 32x64.
// B k-slices (64x32 bf16 = 4KB) are async-DMA'd to LDS once per block and
// double-buffered; shared by all 4 waves. A fragments load directly.
__global__ void __launch_bounds__(128)
gemm_bf16_nt(const bf16_t* __restrict__ A, const bf16_t* __restrict__ B,
             const float* __restrict__ bias, void* __restrict__ Out,
             int M, int N, int K, int out_bf16) {
  const int tid  = threadIdx.x;
  const int lane = tid & 31;
  const int wave = tid >> 5;
  const int m0   = blockIdx.y * 128 + wave * 32;
  const int n0   = blockIdx.x * 64;
  const int half = lane >> 4;
  const int l15  = lane & 15;
  const int k0   = half * 8;

  __shared__ __align__(16) bf16_t Bs[2][64 * 32];   // [buf][n_local*32 + k]

  // Stage one 64x32 B slice into Bs[buf]: 256 16B chunks over 128 threads.
  auto stage = [&](int kk, int buf) {
#pragma unroll
    for (int it = 0; it < 2; ++it) {
      const int c  = tid + it * 128;          // chunk id 0..255
      const int nl = c >> 2;                  // local row 0..63
      const int ko = (c & 3) * 8;             // k element offset 0..24
      async_copy_b128(B + (size_t)(n0 + nl) * K + kk + ko,
                      &Bs[buf][nl * 32 + ko]);
    }
  };

  const bf16_t* arow0 = A + (size_t)(m0 + l15) * K + k0;        // rows m0..m0+15
  const bf16_t* arow1 = arow0 + (size_t)16 * K;                 // rows m0+16..m0+31

  v8f acc0[4] = {{}, {}, {}, {}};
  v8f acc1[4] = {{}, {}, {}, {}};

  stage(0, 0);
  int buf = 0;
  for (int kk = 0; kk < K; kk += 32, buf ^= 1) {
    const bool has_next = (kk + 32) < K;
    if (has_next) {
      stage(kk + 32, buf ^ 1);
      wait_asynccnt_le2();      // current slice's 2 chunks/thread completed
    } else {
      wait_asynccnt_0();
    }
    __syncthreads();            // all waves' DMA chunks visible

    __builtin_prefetch(arow0 + kk + 128, 0, 1);
    __builtin_prefetch(arow1 + kk + 128, 0, 1);
    const v16bf a0 = load_frag(arow0 + kk);
    const v16bf a1 = load_frag(arow1 + kk);
#pragma unroll
    for (int j = 0; j < 4; ++j) {
      const v16bf b = load_frag(&Bs[buf][(j * 16 + l15) * 32 + k0]);
      acc0[j] = wmma_bf16(a0, b, acc0[j]);
      acc1[j] = wmma_bf16(a1, b, acc1[j]);
    }
    __syncthreads();            // buffer may be overwritten next iteration
  }

#pragma unroll
  for (int j = 0; j < 4; ++j) {
    const int col = n0 + j * 16 + l15;
    const float bv = bias[col];
#pragma unroll
    for (int r = 0; r < 8; ++r) {
      const int row0 = m0 + half * 8 + r;          // C/D layout: VGPR r -> M=r+8*half
      const int row1 = row0 + 16;
      const float v0 = acc0[j][r] + bv;
      const float v1 = acc1[j][r] + bv;
      if (out_bf16) {
        ((bf16_t*)Out)[(size_t)row0 * N + col] = (bf16_t)v0;
        ((bf16_t*)Out)[(size_t)row1 * N + col] = (bf16_t)v1;
      } else {
        ((float*)Out)[(size_t)row0 * N + col] = v0;
        ((float*)Out)[(size_t)row1 * N + col] = v1;
      }
    }
  }
}

// ---------------------------------------------------------------- attention
// One wave per (head, 16-query tile). Keys span [t0-128, t0+15] = 9 tiles of 16.
__global__ void __launch_bounds__(32)
attn_kernel(const bf16_t* __restrict__ Q, const bf16_t* __restrict__ Kmat,
            const bf16_t* __restrict__ V, bf16_t* __restrict__ Ctx) {
  const int h    = blockIdx.x;
  const int t0   = blockIdx.y * 16;
  const int kb   = t0 - 128;           // first key in the staged range
  const int lane = threadIdx.x;
  const int half = lane >> 4;
  const int l15  = lane & 15;
  const int k0   = half * 8;

  __shared__ __align__(16) float  S[16][160];   // scores (cols 0..143 used)
  __shared__ __align__(16) bf16_t P[16][160];   // probs, zero-padded to K=160
  __shared__ __align__(16) bf16_t Vt[64][160];  // V transposed: [d][w]

  // Stage transposed V window; zero rows for t<0 (reference zeroes vw) and pad.
  for (int w = 0; w < 160; ++w) {
    const int key = kb + w;
    const bool ok = (w < 144) && (key >= 0);
    const bf16_t* vrow = V + (size_t)key * C_DIM + h * D_HEAD;
    for (int d = lane; d < D_HEAD; d += 32)
      Vt[d][w] = ok ? vrow[d] : (bf16_t)0.0f;
  }
  __syncthreads();

  // ---- S = (Q . K^T) * 1/sqrt(D); zero-K fragments reproduce "score 0" for t<0
  const bf16_t* qrow = Q + (size_t)(t0 + l15) * C_DIM + h * D_HEAD + k0;
  const v16bf a0 = load_frag(qrow);        // d = 0..31 slice
  const v16bf a1 = load_frag(qrow + 32);   // d = 32..63 slice
  for (int j = 0; j < 9; ++j) {
    const int key = kb + j * 16 + l15;     // same key for both wave halves
    v16bf b0, b1;
    if (key >= 0) {
      const bf16_t* krow = Kmat + (size_t)key * C_DIM + h * D_HEAD + k0;
      b0 = load_frag(krow);
      b1 = load_frag(krow + 32);
    } else {
      v8bf z = zero8();
      b0 = cat16(z, z);
      b1 = cat16(z, z);
    }
    v8f s = {};
    s = wmma_bf16(a0, b0, s);
    s = wmma_bf16(a1, b1, s);
#pragma unroll
    for (int r = 0; r < 8; ++r)
      S[r + half * 8][j * 16 + l15] = s[r] * 0.125f;  // 1/sqrt(64)
  }
  __syncthreads();

  // ---- row softmax over the window band: row r uses columns [r+1, r+128]
  if (lane < 16) {
    const int r = lane;
    float mx = -1e30f;
    for (int c = r + 1; c <= r + 128; ++c) mx = fmaxf(mx, S[r][c]);
    float sum = 0.0f;
    for (int c = r + 1; c <= r + 128; ++c) {
      const float e = __expf(S[r][c] - mx);
      S[r][c] = e;
      sum += e;
    }
    const float inv = 1.0f / sum;
    for (int c = 0; c < 160; ++c) {
      const bool in = (c >= r + 1) && (c <= r + 128);
      P[r][c] = in ? (bf16_t)(S[r][c] * inv) : (bf16_t)0.0f;
    }
  }
  __syncthreads();

  // ---- ctx = P (16x160) . Vt^T  -> 16x64, K-loop of 5 from LDS
  v8f cacc[4] = {{}, {}, {}, {}};
  const bf16_t* prow = &P[l15][0] + k0;
  for (int kk = 0; kk < 160; kk += 32) {
    const v16bf a = load_frag(prow + kk);
#pragma unroll
    for (int j = 0; j < 4; ++j) {
      const v16bf b = load_frag(&Vt[j * 16 + l15][kk + k0]);
      cacc[j] = wmma_bf16(a, b, cacc[j]);
    }
  }
#pragma unroll
  for (int j = 0; j < 4; ++j)
#pragma unroll
    for (int r = 0; r < 8; ++r) {
      const int row = t0 + half * 8 + r;
      const int col = h * D_HEAD + j * 16 + l15;
      Ctx[(size_t)row * C_DIM + col] = (bf16_t)cacc[j][r];
    }
}

// ---------------------------------------------------------------- launch
extern "C" void kernel_launch(void* const* d_in, const int* in_sizes, int n_in,
                              void* d_out, int out_size, void* d_ws, size_t ws_size,
                              hipStream_t stream) {
  (void)in_sizes; (void)n_in; (void)out_size; (void)ws_size;
  const float* x  = (const float*)d_in[0];
  const float* wq = (const float*)d_in[1];
  const float* bq = (const float*)d_in[2];
  const float* wk = (const float*)d_in[3];
  const float* bk = (const float*)d_in[4];
  const float* wv = (const float*)d_in[5];
  const float* bv = (const float*)d_in[6];
  const float* wo = (const float*)d_in[7];
  const float* bo = (const float*)d_in[8];
  float* out = (float*)d_out;

  const int nTC = T_SEQ * C_DIM;      // 2M elems
  const int nCC = C_DIM * C_DIM;      // 1M elems

  char* ws = (char*)d_ws;
  size_t off = 0;
  bf16_t* xb   = (bf16_t*)(ws + off); off += (size_t)nTC * 2;
  bf16_t* wqb  = (bf16_t*)(ws + off); off += (size_t)nCC * 2;
  bf16_t* wkb  = (bf16_t*)(ws + off); off += (size_t)nCC * 2;
  bf16_t* wvb  = (bf16_t*)(ws + off); off += (size_t)nCC * 2;
  bf16_t* wob  = (bf16_t*)(ws + off); off += (size_t)nCC * 2;
  bf16_t* qb   = (bf16_t*)(ws + off); off += (size_t)nTC * 2;
  bf16_t* kbuf = (bf16_t*)(ws + off); off += (size_t)nTC * 2;
  bf16_t* vb   = (bf16_t*)(ws + off); off += (size_t)nTC * 2;
  bf16_t* ctxb = (bf16_t*)(ws + off); off += (size_t)nTC * 2;

  f32_to_bf16_kernel<<<(nTC / 8 + 255) / 256, 256, 0, stream>>>(x,  xb,  nTC);
  f32_to_bf16_kernel<<<(nCC / 8 + 255) / 256, 256, 0, stream>>>(wq, wqb, nCC);
  f32_to_bf16_kernel<<<(nCC / 8 + 255) / 256, 256, 0, stream>>>(wk, wkb, nCC);
  f32_to_bf16_kernel<<<(nCC / 8 + 255) / 256, 256, 0, stream>>>(wv, wvb, nCC);
  f32_to_bf16_kernel<<<(nCC / 8 + 255) / 256, 256, 0, stream>>>(wo, wob, nCC);

  dim3 gg(C_DIM / 64, T_SEQ / 128);   // (16, 16)
  gemm_bf16_nt<<<gg, 128, 0, stream>>>(xb, wqb, bq, qb,   T_SEQ, C_DIM, C_DIM, 1);
  gemm_bf16_nt<<<gg, 128, 0, stream>>>(xb, wkb, bk, kbuf, T_SEQ, C_DIM, C_DIM, 1);
  gemm_bf16_nt<<<gg, 128, 0, stream>>>(xb, wvb, bv, vb,   T_SEQ, C_DIM, C_DIM, 1);

  attn_kernel<<<dim3(H_NUM, T_SEQ / 16), 32, 0, stream>>>(qb, kbuf, vb, ctxb);

  gemm_bf16_nt<<<gg, 128, 0, stream>>>(ctxb, wob, bo, out, T_SEQ, C_DIM, C_DIM, 0);
}
